// GCN_5755256177073
// MI455X (gfx1250) — matrix operand order repaired and checked by reading, not compiled, not run on hardware
//
#include <hip/hip_runtime.h>

#define NN    10000
#define EE    320000
#define E2    (EE + NN)
#define DIN   512
#define HH    256
#define HEADS 4
#define DH    64
#define GG    64
#define OUTD  512

typedef __attribute__((ext_vector_type(16))) __bf16 v16bf;
typedef __attribute__((ext_vector_type(8)))  float  v8f;

#define WMMA_BF16(a, b, c) \
  __builtin_amdgcn_wmma_f32_16x16x32_bf16(false, (a), false, (b), (short)0, (c), false, false)

__device__ __forceinline__ float gelu_f(float x) {
  return 0.5f * x * (1.0f + erff(x * 0.70710678118654752f));
}
__device__ __forceinline__ void split2(float x, __bf16& h, __bf16& l) {
  h = (__bf16)x;
  l = (__bf16)(x - (float)h);
}
// monotonic float -> uint encoding for atomicMax-based segment_max (0u acts as -inf)
__device__ __forceinline__ unsigned fenc(float f) {
  unsigned u = __float_as_uint(f);
  return ((int)u >= 0) ? (u | 0x80000000u) : ~u;
}
__device__ __forceinline__ float fdec(unsigned k) {
  unsigned u = (k & 0x80000000u) ? (k & 0x7fffffffu) : ~k;
  return __uint_as_float(u);
}

// A tile (16xK, bf16) from LDS, CDNA5 layout: lane<16 row=lane, K=e(+8 for e>=8); lane>=16 +8
__device__ __forceinline__ v16bf load_a_lds(const __bf16* As, int ldk, int k0, int lane) {
  const __bf16* p = As + (lane & 15) * ldk + k0 + ((lane >> 4) << 3);
  v16bf a;
#pragma unroll
  for (int e = 0; e < 16; ++e) a[e] = p[(e < 8) ? e : (e + 8)];
  return a;
}

// B tile (32x16) from global f32 weights, split to bf16 hi/lo on the fly.
// lane layout: N = lane&15, K = 8*(lane>>4) + e (+8 for e>=8)
__device__ __forceinline__ void load_b_split(const float* W, int ldn, int k0, int n0,
                                             int lane, v16bf& bh, v16bf& bl) {
  const float* p = W + (size_t)(k0 + ((lane >> 4) << 3)) * ldn + n0 + (lane & 15);
#pragma unroll
  for (int e = 0; e < 16; ++e) {
    float w = p[(size_t)((e < 8) ? e : (e + 8)) * ldn];
    __bf16 h = (__bf16)w;
    bh[e] = h;
    bl[e] = (__bf16)(w - (float)h);
  }
}

__global__ void k_zero(float* p, size_t n) {
  size_t i = (size_t)blockIdx.x * blockDim.x + threadIdx.x;
  size_t stride = (size_t)gridDim.x * blockDim.x;
  for (; i < n; i += stride) p[i] = 0.0f;
}

// ---------------- K1: x_res = LN(gelu(LN(x) @ Wp + bp)) -------------------
__global__ __launch_bounds__(256) void k_ln_proj(
    const float* __restrict__ x, const float* __restrict__ g_in, const float* __restrict__ b_in,
    const float* __restrict__ Wp, const float* __restrict__ bp,
    const float* __restrict__ g_p, const float* __restrict__ b_p,
    float* __restrict__ x_res) {
  __shared__ float  xs[16][DIN + 4];
  __shared__ __bf16 ah[16][DIN + 8];
  __shared__ __bf16 al[16][DIN + 8];
  __shared__ float  cs[16][HH + 4];
  __shared__ float  red[16][16];
  __shared__ float  stat_m[16], stat_r[16];

  const int t = threadIdx.x;
  const int row0 = blockIdx.x * 16;
  const int r = t >> 4, c16 = t & 15;

  for (int i = t; i < 16 * DIN; i += 256) {
    int rr = i >> 9, cc = i & (DIN - 1);
    xs[rr][cc] = x[(size_t)(row0 + rr) * DIN + cc];
  }
  __syncthreads();

  float s0 = 0.f, s1 = 0.f;
  for (int c = c16; c < DIN; c += 16) { float v = xs[r][c]; s0 += v; s1 += v * v; }
  red[r][c16] = s0;
  __syncthreads();
  if (t < 16) { float a = 0.f; for (int j = 0; j < 16; ++j) a += red[t][j]; stat_m[t] = a * (1.0f / DIN); }
  __syncthreads();
  red[r][c16] = s1;
  __syncthreads();
  if (t < 16) {
    float a = 0.f; for (int j = 0; j < 16; ++j) a += red[t][j];
    float m = stat_m[t];
    stat_r[t] = rsqrtf(a * (1.0f / DIN) - m * m + 1e-5f);
  }
  __syncthreads();
  {
    float m = stat_m[r], rs = stat_r[r];
    for (int c = c16; c < DIN; c += 16) {
      float v = (xs[r][c] - m) * rs * g_in[c] + b_in[c];
      split2(v, ah[r][c], al[r][c]);
    }
  }
  __syncthreads();

  const int w = t >> 5, lane = t & 31, n0 = w * 32;
  v8f acc0 = {}, acc1 = {};
  for (int k0 = 0; k0 < DIN; k0 += 32) {
    v16bf A_h = load_a_lds(&ah[0][0], DIN + 8, k0, lane);
    v16bf A_l = load_a_lds(&al[0][0], DIN + 8, k0, lane);
    v16bf bh, bl;
    load_b_split(Wp, HH, k0, n0, lane, bh, bl);
    acc0 = WMMA_BF16(A_h, bh, acc0);
    acc0 = WMMA_BF16(A_h, bl, acc0);
    acc0 = WMMA_BF16(A_l, bh, acc0);
    load_b_split(Wp, HH, k0, n0 + 16, lane, bh, bl);
    acc1 = WMMA_BF16(A_h, bh, acc1);
    acc1 = WMMA_BF16(A_h, bl, acc1);
    acc1 = WMMA_BF16(A_l, bh, acc1);
  }
  {
    int col = lane & 15, mb = (lane >> 4) << 3;
#pragma unroll
    for (int q = 0; q < 8; ++q) {
      cs[mb + q][n0 + col]      = gelu_f(acc0[q] + bp[n0 + col]);
      cs[mb + q][n0 + 16 + col] = gelu_f(acc1[q] + bp[n0 + 16 + col]);
    }
  }
  __syncthreads();

  s0 = 0.f; s1 = 0.f;
  for (int c = c16; c < HH; c += 16) { float v = cs[r][c]; s0 += v; s1 += v * v; }
  red[r][c16] = s0;
  __syncthreads();
  if (t < 16) { float a = 0.f; for (int j = 0; j < 16; ++j) a += red[t][j]; stat_m[t] = a * (1.0f / HH); }
  __syncthreads();
  red[r][c16] = s1;
  __syncthreads();
  if (t < 16) {
    float a = 0.f; for (int j = 0; j < 16; ++j) a += red[t][j];
    float m = stat_m[t];
    stat_r[t] = rsqrtf(a * (1.0f / HH) - m * m + 1e-5f);
  }
  __syncthreads();
  {
    float m = stat_m[r], rs = stat_r[r];
    for (int c = c16; c < HH; c += 16)
      x_res[(size_t)(row0 + r) * HH + c] = (cs[r][c] - m) * rs * g_p[c] + b_p[c];
  }
}

// ---------------- K2: agg[dst] += x_res[src]  (L2-resident atomics) -------
__global__ void k_agg_scatter(const int* __restrict__ src, const int* __restrict__ dst,
                              const float* __restrict__ x_res, float* __restrict__ agg) {
  size_t idx = (size_t)blockIdx.x * blockDim.x + threadIdx.x;
  if (idx >= (size_t)EE * 64) return;
  int e = (int)(idx >> 6), j = (int)(idx & 63);
  int s = src[e], d = dst[e];
  float4 v = ((const float4*)x_res)[(size_t)s * 64 + j];
  float* p = agg + (size_t)d * HH + j * 4;
  atomicAdd(p + 0, v.x); atomicAdd(p + 1, v.y);
  atomicAdd(p + 2, v.z); atomicAdd(p + 3, v.w);
}

// ------- K3: x1 = LN(gelu(agg@Wrel + brel + x_res@Wroot)) + x_res ---------
__global__ __launch_bounds__(256) void k_graphconv(
    const float* __restrict__ agg, const float* __restrict__ x_res,
    const float* __restrict__ Wrel, const float* __restrict__ brel,
    const float* __restrict__ Wroot,
    const float* __restrict__ n1g, const float* __restrict__ n1b,
    float* __restrict__ x1) {
  __shared__ __bf16 ah_a[16][HH + 8], al_a[16][HH + 8];
  __shared__ __bf16 ah_r[16][HH + 8], al_r[16][HH + 8];
  __shared__ float  rs_f[16][HH + 4];
  __shared__ float  cs[16][HH + 4];
  __shared__ float  red[16][16];
  __shared__ float  stat_m[16], stat_r[16];

  const int t = threadIdx.x;
  const int row0 = blockIdx.x * 16;
  const int r = t >> 4, c16 = t & 15;

  for (int i = t; i < 16 * HH; i += 256) {
    int rr = i >> 8, cc = i & (HH - 1);
    float va = agg[(size_t)(row0 + rr) * HH + cc];
    float vr = x_res[(size_t)(row0 + rr) * HH + cc];
    split2(va, ah_a[rr][cc], al_a[rr][cc]);
    split2(vr, ah_r[rr][cc], al_r[rr][cc]);
    rs_f[rr][cc] = vr;
  }
  __syncthreads();

  const int w = t >> 5, lane = t & 31, n0 = w * 32;
  v8f acc0 = {}, acc1 = {};
  for (int k0 = 0; k0 < HH; k0 += 32) {
    v16bf Aah = load_a_lds(&ah_a[0][0], HH + 8, k0, lane);
    v16bf Aal = load_a_lds(&al_a[0][0], HH + 8, k0, lane);
    v16bf Arh = load_a_lds(&ah_r[0][0], HH + 8, k0, lane);
    v16bf Arl = load_a_lds(&al_r[0][0], HH + 8, k0, lane);
    v16bf bh, bl;
    load_b_split(Wrel, HH, k0, n0, lane, bh, bl);
    acc0 = WMMA_BF16(Aah, bh, acc0); acc0 = WMMA_BF16(Aah, bl, acc0); acc0 = WMMA_BF16(Aal, bh, acc0);
    load_b_split(Wroot, HH, k0, n0, lane, bh, bl);
    acc0 = WMMA_BF16(Arh, bh, acc0); acc0 = WMMA_BF16(Arh, bl, acc0); acc0 = WMMA_BF16(Arl, bh, acc0);
    load_b_split(Wrel, HH, k0, n0 + 16, lane, bh, bl);
    acc1 = WMMA_BF16(Aah, bh, acc1); acc1 = WMMA_BF16(Aah, bl, acc1); acc1 = WMMA_BF16(Aal, bh, acc1);
    load_b_split(Wroot, HH, k0, n0 + 16, lane, bh, bl);
    acc1 = WMMA_BF16(Arh, bh, acc1); acc1 = WMMA_BF16(Arh, bl, acc1); acc1 = WMMA_BF16(Arl, bh, acc1);
  }
  {
    int col = lane & 15, mb = (lane >> 4) << 3;
#pragma unroll
    for (int q = 0; q < 8; ++q) {
      cs[mb + q][n0 + col]      = gelu_f(acc0[q] + brel[n0 + col]);
      cs[mb + q][n0 + 16 + col] = gelu_f(acc1[q] + brel[n0 + 16 + col]);
    }
  }
  __syncthreads();

  float s0 = 0.f, s1 = 0.f;
  for (int c = c16; c < HH; c += 16) { float v = cs[r][c]; s0 += v; s1 += v * v; }
  red[r][c16] = s0;
  __syncthreads();
  if (t < 16) { float a = 0.f; for (int j = 0; j < 16; ++j) a += red[t][j]; stat_m[t] = a * (1.0f / HH); }
  __syncthreads();
  red[r][c16] = s1;
  __syncthreads();
  if (t < 16) {
    float a = 0.f; for (int j = 0; j < 16; ++j) a += red[t][j];
    float m = stat_m[t];
    stat_r[t] = rsqrtf(a * (1.0f / HH) - m * m + 1e-5f);
  }
  __syncthreads();
  {
    float m = stat_m[r], rs = stat_r[r];
    for (int c = c16; c < HH; c += 16)
      x1[(size_t)(row0 + r) * HH + c] = (cs[r][c] - m) * rs * n1g[c] + n1b[c] + rs_f[r][c];
  }
}

// ---------------- K4: xl = x1@Wl + bl ; xr = x1@Wr + br -------------------
__global__ __launch_bounds__(256) void k_gat_lr(
    const float* __restrict__ x1,
    const float* __restrict__ Wl, const float* __restrict__ bl,
    const float* __restrict__ Wr, const float* __restrict__ br,
    float* __restrict__ xl, float* __restrict__ xr) {
  __shared__ __bf16 ah[16][HH + 8], al[16][HH + 8];
  const int t = threadIdx.x;
  const int row0 = blockIdx.x * 16;
  for (int i = t; i < 16 * HH; i += 256) {
    int rr = i >> 8, cc = i & (HH - 1);
    split2(x1[(size_t)(row0 + rr) * HH + cc], ah[rr][cc], al[rr][cc]);
  }
  __syncthreads();

  const int w = t >> 5, lane = t & 31, n0 = w * 32;
  const int col = lane & 15, mb = (lane >> 4) << 3;
  {
    v8f a0 = {}, a1 = {};
    for (int k0 = 0; k0 < HH; k0 += 32) {
      v16bf Ah = load_a_lds(&ah[0][0], HH + 8, k0, lane);
      v16bf Al = load_a_lds(&al[0][0], HH + 8, k0, lane);
      v16bf bhv, blv;
      load_b_split(Wl, HH, k0, n0, lane, bhv, blv);
      a0 = WMMA_BF16(Ah, bhv, a0); a0 = WMMA_BF16(Ah, blv, a0); a0 = WMMA_BF16(Al, bhv, a0);
      load_b_split(Wl, HH, k0, n0 + 16, lane, bhv, blv);
      a1 = WMMA_BF16(Ah, bhv, a1); a1 = WMMA_BF16(Ah, blv, a1); a1 = WMMA_BF16(Al, bhv, a1);
    }
#pragma unroll
    for (int q = 0; q < 8; ++q) {
      xl[(size_t)(row0 + mb + q) * HH + n0 + col]      = a0[q] + bl[n0 + col];
      xl[(size_t)(row0 + mb + q) * HH + n0 + 16 + col] = a1[q] + bl[n0 + 16 + col];
    }
  }
  {
    v8f a0 = {}, a1 = {};
    for (int k0 = 0; k0 < HH; k0 += 32) {
      v16bf Ah = load_a_lds(&ah[0][0], HH + 8, k0, lane);
      v16bf Al = load_a_lds(&al[0][0], HH + 8, k0, lane);
      v16bf bhv, blv;
      load_b_split(Wr, HH, k0, n0, lane, bhv, blv);
      a0 = WMMA_BF16(Ah, bhv, a0); a0 = WMMA_BF16(Ah, blv, a0); a0 = WMMA_BF16(Al, bhv, a0);
      load_b_split(Wr, HH, k0, n0 + 16, lane, bhv, blv);
      a1 = WMMA_BF16(Ah, bhv, a1); a1 = WMMA_BF16(Ah, blv, a1); a1 = WMMA_BF16(Al, bhv, a1);
    }
#pragma unroll
    for (int q = 0; q < 8; ++q) {
      xr[(size_t)(row0 + mb + q) * HH + n0 + col]      = a0[q] + br[n0 + col];
      xr[(size_t)(row0 + mb + q) * HH + n0 + 16 + col] = a1[q] + br[n0 + 16 + col];
    }
  }
}

// ---------------- K5: GAT logits + segment max ----------------------------
__global__ void k_gat_logits(const int* __restrict__ src, const int* __restrict__ dst,
                             const float* __restrict__ xl, const float* __restrict__ xr,
                             const float* __restrict__ att,
                             float* __restrict__ logit, unsigned* __restrict__ nmax) {
  size_t idx = (size_t)blockIdx.x * blockDim.x + threadIdx.x;
  if (idx >= (size_t)E2 * HEADS) return;
  int e = (int)(idx >> 2), h = (int)(idx & 3);
  int s = (e < EE) ? src[e] : (e - EE);
  int d = (e < EE) ? dst[e] : (e - EE);
  const float* pl = xl + (size_t)s * HH + h * DH;
  const float* pr = xr + (size_t)d * HH + h * DH;
  const float* pa = att + h * DH;
  float acc = 0.f;
  for (int k = 0; k < DH; ++k) {
    float v = pl[k] + pr[k];
    v = (v > 0.f) ? v : 0.2f * v;
    acc += v * pa[k];
  }
  logit[idx] = acc;
  atomicMax(nmax + (size_t)d * HEADS + h, fenc(acc));
}

// ---------------- K6: a = exp(logit - max); den[d] += a -------------------
__global__ void k_gat_soft(const int* __restrict__ dst, const unsigned* __restrict__ nmax,
                           float* __restrict__ logit, float* __restrict__ den) {
  size_t idx = (size_t)blockIdx.x * blockDim.x + threadIdx.x;
  if (idx >= (size_t)E2 * HEADS) return;
  int e = (int)(idx >> 2), h = (int)(idx & 3);
  int d = (e < EE) ? dst[e] : (e - EE);
  float a = expf(logit[idx] - fdec(nmax[(size_t)d * HEADS + h]));
  logit[idx] = a;
  atomicAdd(den + (size_t)d * HEADS + h, a);
}

// ---------------- K7: x2[d] += (a/den) * xl[s] ----------------------------
__global__ void k_gat_msg(const int* __restrict__ src, const int* __restrict__ dst,
                          const float* __restrict__ xl, const float* __restrict__ logit,
                          const float* __restrict__ den, float* __restrict__ x2) {
  size_t idx = (size_t)blockIdx.x * blockDim.x + threadIdx.x;
  if (idx >= (size_t)E2 * 64) return;
  int e = (int)(idx >> 6), j = (int)(idx & 63), h = j >> 4;
  int s = (e < EE) ? src[e] : (e - EE);
  int d = (e < EE) ? dst[e] : (e - EE);
  float alpha = logit[(size_t)e * HEADS + h] / den[(size_t)d * HEADS + h];
  float4 v = ((const float4*)xl)[(size_t)s * 64 + j];
  float* p = x2 + (size_t)d * HH + j * 4;
  atomicAdd(p + 0, alpha * v.x); atomicAdd(p + 1, alpha * v.y);
  atomicAdd(p + 2, alpha * v.z); atomicAdd(p + 3, alpha * v.w);
}

// ---------------- K8: x2 = LN(gelu(x2 + gat_b)) + x1 ----------------------
__global__ __launch_bounds__(256) void k_x2_post(
    const float* __restrict__ gat_b, const float* __restrict__ n2g, const float* __restrict__ n2b,
    const float* __restrict__ x1, float* __restrict__ x2) {
  __shared__ float cs[16][HH + 4];
  __shared__ float red[16][16];
  __shared__ float stat_m[16], stat_r[16];
  const int t = threadIdx.x;
  const int row0 = blockIdx.x * 16;
  const int r = t >> 4, c16 = t & 15;
  for (int i = t; i < 16 * HH; i += 256) {
    int rr = i >> 8, cc = i & (HH - 1);
    cs[rr][cc] = gelu_f(x2[(size_t)(row0 + rr) * HH + cc] + gat_b[cc]);
  }
  __syncthreads();
  float s0 = 0.f, s1 = 0.f;
  for (int c = c16; c < HH; c += 16) { float v = cs[r][c]; s0 += v; s1 += v * v; }
  red[r][c16] = s0;
  __syncthreads();
  if (t < 16) { float a = 0.f; for (int j = 0; j < 16; ++j) a += red[t][j]; stat_m[t] = a * (1.0f / HH); }
  __syncthreads();
  red[r][c16] = s1;
  __syncthreads();
  if (t < 16) {
    float a = 0.f; for (int j = 0; j < 16; ++j) a += red[t][j];
    float m = stat_m[t];
    stat_r[t] = rsqrtf(a * (1.0f / HH) - m * m + 1e-5f);
  }
  __syncthreads();
  {
    float m = stat_m[r], rs = stat_r[r];
    for (int c = c16; c < HH; c += 16) {
      size_t o = (size_t)(row0 + r) * HH + c;
      x2[o] = (cs[r][c] - m) * rs * n2g[c] + n2b[c] + x1[o];
    }
  }
}

// ------- K9: gates (two WMMA GEMMs + tanh + dot) + group max --------------
__global__ __launch_bounds__(256) void k_gates(
    const float* __restrict__ x2, const int* __restrict__ batch,
    const float* __restrict__ Wg1a, const float* __restrict__ bg1a,
    const float* __restrict__ Wg1b, const float* __restrict__ bg1b,
    const float* __restrict__ Wg2a, const float* __restrict__ bg2a,
    const float* __restrict__ Wg2b, const float* __restrict__ bg2b,
    float* __restrict__ gate1, float* __restrict__ gate2,
    unsigned* __restrict__ gm1, unsigned* __restrict__ gm2) {
  __shared__ __bf16 ah[16][HH + 8], al[16][HH + 8];
  __shared__ float  t1[16][128 + 4];
  __shared__ float  t2[16][HH + 4];
  __shared__ float  red[16][16];
  const int t = threadIdx.x;
  const int row0 = blockIdx.x * 16;
  for (int i = t; i < 16 * HH; i += 256) {
    int rr = i >> 8, cc = i & (HH - 1);
    split2(x2[(size_t)(row0 + rr) * HH + cc], ah[rr][cc], al[rr][cc]);
  }
  __syncthreads();

  const int w = t >> 5, lane = t & 31;
  const int col = lane & 15, mb = (lane >> 4) << 3;
  { // gate1 hidden: N=128, one 16-wide tile per wave
    int n0 = w * 16;
    v8f a = {};
    for (int k0 = 0; k0 < HH; k0 += 32) {
      v16bf Ah = load_a_lds(&ah[0][0], HH + 8, k0, lane);
      v16bf Al = load_a_lds(&al[0][0], HH + 8, k0, lane);
      v16bf bhv, blv;
      load_b_split(Wg1a, 128, k0, n0, lane, bhv, blv);
      a = WMMA_BF16(Ah, bhv, a); a = WMMA_BF16(Ah, blv, a); a = WMMA_BF16(Al, bhv, a);
    }
#pragma unroll
    for (int q = 0; q < 8; ++q) t1[mb + q][n0 + col] = tanhf(a[q] + bg1a[n0 + col]);
  }
  { // gate2 hidden: N=256, two tiles per wave
    int n0 = w * 32;
    v8f a0 = {}, a1 = {};
    for (int k0 = 0; k0 < HH; k0 += 32) {
      v16bf Ah = load_a_lds(&ah[0][0], HH + 8, k0, lane);
      v16bf Al = load_a_lds(&al[0][0], HH + 8, k0, lane);
      v16bf bhv, blv;
      load_b_split(Wg2a, HH, k0, n0, lane, bhv, blv);
      a0 = WMMA_BF16(Ah, bhv, a0); a0 = WMMA_BF16(Ah, blv, a0); a0 = WMMA_BF16(Al, bhv, a0);
      load_b_split(Wg2a, HH, k0, n0 + 16, lane, bhv, blv);
      a1 = WMMA_BF16(Ah, bhv, a1); a1 = WMMA_BF16(Ah, blv, a1); a1 = WMMA_BF16(Al, bhv, a1);
    }
#pragma unroll
    for (int q = 0; q < 8; ++q) {
      t2[mb + q][n0 + col]      = tanhf(a0[q] + bg2a[n0 + col]);
      t2[mb + q][n0 + 16 + col] = tanhf(a1[q] + bg2a[n0 + 16 + col]);
    }
  }
  __syncthreads();

  const int r = t >> 4, c16 = t & 15;
  float s = 0.f;
  for (int j = c16; j < 128; j += 16) s += t1[r][j] * Wg1b[j];
  red[r][c16] = s;
  __syncthreads();
  if (t < 16) {
    float a = 0.f; for (int j = 0; j < 16; ++j) a += red[t][j];
    float v = a + bg1b[0];
    int node = row0 + t;
    gate1[node] = v;
    atomicMax(gm1 + batch[node], fenc(v));
  }
  __syncthreads();
  s = 0.f;
  for (int j = c16; j < HH; j += 16) s += t2[r][j] * Wg2b[j];
  red[r][c16] = s;
  __syncthreads();
  if (t < 16) {
    float a = 0.f; for (int j = 0; j < 16; ++j) a += red[t][j];
    float v = a + bg2b[0];
    int node = row0 + t;
    gate2[node] = v;
    atomicMax(gm2 + batch[node], fenc(v));
  }
}

// ---------------- K10: gate exp + group sums + counts ---------------------
__global__ void k_gate_norm(const int* __restrict__ batch,
                            const unsigned* __restrict__ gm1, const unsigned* __restrict__ gm2,
                            float* __restrict__ gate1, float* __restrict__ gate2,
                            float* __restrict__ gs1, float* __restrict__ gs2,
                            float* __restrict__ cnt) {
  int n = blockIdx.x * blockDim.x + threadIdx.x;
  if (n >= NN) return;
  int b = batch[n];
  float e1 = expf(gate1[n] - fdec(gm1[b])); gate1[n] = e1; atomicAdd(gs1 + b, e1);
  float e2 = expf(gate2[n] - fdec(gm2[b])); gate2[n] = e2; atomicAdd(gs2 + b, e2);
  atomicAdd(cnt + b, 1.0f);
}

// ---------------- K11: pooled accumulations -------------------------------
__global__ void k_pool(const int* __restrict__ batch,
                       const float* __restrict__ x2, const float* __restrict__ x_res,
                       const float* __restrict__ gate1, const float* __restrict__ gate2,
                       const float* __restrict__ gs1, const float* __restrict__ gs2,
                       float* __restrict__ emb1, float* __restrict__ emb2,
                       float* __restrict__ gres) {
  size_t idx = (size_t)blockIdx.x * blockDim.x + threadIdx.x;
  if (idx >= (size_t)NN * 64) return;
  int n = (int)(idx >> 6), j = (int)(idx & 63);
  int b = batch[n];
  float w1 = gate1[n] / gs1[b];
  float w2 = gate2[n] / gs2[b];
  float4 v2 = ((const float4*)x2)[(size_t)n * 64 + j];
  float4 vr = ((const float4*)x_res)[(size_t)n * 64 + j];
  float* p1 = emb1 + (size_t)b * HH + j * 4;
  float* p2 = emb2 + (size_t)b * HH + j * 4;
  float* pr = gres + (size_t)b * HH + j * 4;
  atomicAdd(p1 + 0, w1 * v2.x); atomicAdd(p1 + 1, w1 * v2.y);
  atomicAdd(p1 + 2, w1 * v2.z); atomicAdd(p1 + 3, w1 * v2.w);
  atomicAdd(p2 + 0, w2 * v2.x); atomicAdd(p2 + 1, w2 * v2.y);
  atomicAdd(p2 + 2, w2 * v2.z); atomicAdd(p2 + 3, w2 * v2.w);
  atomicAdd(pr + 0, vr.x); atomicAdd(pr + 1, vr.y);
  atomicAdd(pr + 2, vr.z); atomicAdd(pr + 3, vr.w);
}

// ---------------- K12: out = BN(gelu([emb1,emb2,gres/cnt] @ Wo + bo)) ------
__global__ void k_final(const float* __restrict__ emb1, const float* __restrict__ emb2,
                        const float* __restrict__ gres, const float* __restrict__ cnt,
                        const float* __restrict__ Wo, const float* __restrict__ bo,
                        const float* __restrict__ bng, const float* __restrict__ bnb,
                        float* __restrict__ out) {
  int idx = blockIdx.x * blockDim.x + threadIdx.x;
  if (idx >= GG * OUTD) return;
  int g = idx >> 9, o = idx & (OUTD - 1);
  float inv = 1.0f / fmaxf(cnt[g], 1.0f);
  const float* e1 = emb1 + (size_t)g * HH;
  const float* e2 = emb2 + (size_t)g * HH;
  const float* gr = gres + (size_t)g * HH;
  float acc = bo[o];
  for (int k = 0; k < HH; ++k) acc += e1[k] * Wo[(size_t)k * OUTD + o];
  for (int k = 0; k < HH; ++k) acc += e2[k] * Wo[(size_t)(HH + k) * OUTD + o];
  for (int k = 0; k < HH; ++k) acc += gr[k] * inv * Wo[(size_t)(2 * HH + k) * OUTD + o];
  float v = gelu_f(acc);
  out[idx] = v * (1.0f / sqrtf(1.0f + 1e-5f)) * bng[o] + bnb[o];
}

// ---------------- workspace layout (float elements) -----------------------
static const size_t OFF_XRES = 0;
static const size_t OFF_X1   = 2560000;
static const size_t OFF_XL   = 5120000;
static const size_t OFF_XR   = 7680000;
static const size_t OFF_LOG  = 10240000;   // E2*HEADS = 1,320,000
static const size_t OFF_G1   = 11560000;   // NN
static const size_t OFF_G2   = 11570000;   // NN
static const size_t OFF_ACC  = 11580000;   // accumulators start (zeroed each call)
static const size_t OFF_AGG  = 11580000;   // NN*HH
static const size_t OFF_X2   = 14140000;   // NN*HH
static const size_t OFF_NMAX = 16700000;   // NN*HEADS (uint-encoded)
static const size_t OFF_DEN  = 16740000;   // NN*HEADS
static const size_t OFF_GM1  = 16780000;   // GG (uint-encoded)
static const size_t OFF_GM2  = 16780064;
static const size_t OFF_GS1  = 16780128;
static const size_t OFF_GS2  = 16780192;
static const size_t OFF_CNT  = 16780256;
static const size_t OFF_EMB1 = 16780320;   // GG*HH
static const size_t OFF_EMB2 = 16796704;
static const size_t OFF_GRES = 16813088;
static const size_t WS_TOTAL = 16829472;   // ~67.3 MB

extern "C" void kernel_launch(void* const* d_in, const int* in_sizes, int n_in,
                              void* d_out, int out_size, void* d_ws, size_t ws_size,
                              hipStream_t stream) {
  (void)in_sizes; (void)n_in; (void)out_size; (void)ws_size;
  const float* x       = (const float*)d_in[0];
  const int*   ei      = (const int*)d_in[1];
  const int*   batch   = (const int*)d_in[2];
  const float* ln_in_g = (const float*)d_in[3];
  const float* ln_in_b = (const float*)d_in[4];
  const float* Wp      = (const float*)d_in[5];
  const float* bp      = (const float*)d_in[6];
  const float* lnp_g   = (const float*)d_in[7];
  const float* lnp_b   = (const float*)d_in[8];
  const float* Wrel    = (const float*)d_in[9];
  const float* brel    = (const float*)d_in[10];
  const float* Wroot   = (const float*)d_in[11];
  const float* n1g     = (const float*)d_in[12];
  const float* n1b     = (const float*)d_in[13];
  const float* Wl      = (const float*)d_in[14];
  const float* bl      = (const float*)d_in[15];
  const float* Wr      = (const float*)d_in[16];
  const float* br      = (const float*)d_in[17];
  const float* att     = (const float*)d_in[18];
  const float* gat_b   = (const float*)d_in[19];
  const float* n2g     = (const float*)d_in[20];
  const float* n2b     = (const float*)d_in[21];
  const float* Wg1a    = (const float*)d_in[22];
  const float* bg1a    = (const float*)d_in[23];
  const float* Wg1b    = (const float*)d_in[24];
  const float* bg1b    = (const float*)d_in[25];
  const float* Wg2a    = (const float*)d_in[26];
  const float* bg2a    = (const float*)d_in[27];
  const float* Wg2b    = (const float*)d_in[28];
  const float* bg2b    = (const float*)d_in[29];
  const float* Wo      = (const float*)d_in[30];
  const float* bo      = (const float*)d_in[31];
  const float* bng     = (const float*)d_in[32];
  const float* bnb     = (const float*)d_in[33];

  const int* src = ei;
  const int* dst = ei + EE;

  float* ws    = (float*)d_ws;
  float* x_res = ws + OFF_XRES;
  float* x1    = ws + OFF_X1;
  float* xl    = ws + OFF_XL;
  float* xr    = ws + OFF_XR;
  float* logit = ws + OFF_LOG;
  float* gate1 = ws + OFF_G1;
  float* gate2 = ws + OFF_G2;
  float* agg   = ws + OFF_AGG;
  float* x2    = ws + OFF_X2;
  unsigned* nmax = (unsigned*)(ws + OFF_NMAX);
  float* den   = ws + OFF_DEN;
  unsigned* gm1 = (unsigned*)(ws + OFF_GM1);
  unsigned* gm2 = (unsigned*)(ws + OFF_GM2);
  float* gs1   = ws + OFF_GS1;
  float* gs2   = ws + OFF_GS2;
  float* cnt   = ws + OFF_CNT;
  float* emb1  = ws + OFF_EMB1;
  float* emb2  = ws + OFF_EMB2;
  float* gres  = ws + OFF_GRES;

  k_zero<<<2048, 256, 0, stream>>>(ws + OFF_ACC, WS_TOTAL - OFF_ACC);
  k_ln_proj<<<NN / 16, 256, 0, stream>>>(x, ln_in_g, ln_in_b, Wp, bp, lnp_g, lnp_b, x_res);
  k_agg_scatter<<<(unsigned)(((size_t)EE * 64 + 255) / 256), 256, 0, stream>>>(src, dst, x_res, agg);
  k_graphconv<<<NN / 16, 256, 0, stream>>>(agg, x_res, Wrel, brel, Wroot, n1g, n1b, x1);
  k_gat_lr<<<NN / 16, 256, 0, stream>>>(x1, Wl, bl, Wr, br, xl, xr);
  k_gat_logits<<<(unsigned)(((size_t)E2 * HEADS + 255) / 256), 256, 0, stream>>>(src, dst, xl, xr, att, logit, nmax);
  k_gat_soft<<<(unsigned)(((size_t)E2 * HEADS + 255) / 256), 256, 0, stream>>>(dst, nmax, logit, den);
  k_gat_msg<<<(unsigned)(((size_t)E2 * 64 + 255) / 256), 256, 0, stream>>>(src, dst, xl, logit, den, x2);
  k_x2_post<<<NN / 16, 256, 0, stream>>>(gat_b, n2g, n2b, x1, x2);
  k_gates<<<NN / 16, 256, 0, stream>>>(x2, batch, Wg1a, bg1a, Wg1b, bg1b, Wg2a, bg2a, Wg2b, bg2b,
                                       gate1, gate2, gm1, gm2);
  k_gate_norm<<<(NN + 255) / 256, 256, 0, stream>>>(batch, gm1, gm2, gate1, gate2, gs1, gs2, cnt);
  k_pool<<<(unsigned)(((size_t)NN * 64 + 255) / 256), 256, 0, stream>>>(batch, x2, x_res, gate1, gate2,
                                                                        gs1, gs2, emb1, emb2, gres);
  k_final<<<(GG * OUTD + 255) / 256, 256, 0, stream>>>(emb1, emb2, gres, cnt, Wo, bo, bng, bnb,
                                                       (float*)d_out);
}
